// LATTE_78623671320670
// MI455X (gfx1250) — matrix operand order repaired and checked by reading, not compiled, not run on hardware
//
#include <hip/hip_runtime.h>
#include <hip/hip_bf16.h>

#define NP 40000
#define NA 20000
#define DD 256
#define EC 400000
#define EW 300000

typedef __attribute__((ext_vector_type(16))) _Float16 v16h;
typedef __attribute__((ext_vector_type(8)))  float    v8f;

// ---------------------------------------------------------------- helpers
__device__ __forceinline__ float dot4f(float4 a, float4 b) {
    return a.x * b.x + a.y * b.y + a.z * b.z + a.w * b.w;
}
// wave32 butterfly reduction: every lane ends with the full sum
__device__ __forceinline__ float wredsum(float v) {
#pragma unroll
    for (int o = 16; o > 0; o >>= 1) v += __shfl_xor(v, o, 32);
    return v;
}
// order-preserving float->uint key for atomicMax-based segment max
__device__ __forceinline__ unsigned f2key(float f) {
    unsigned u = __float_as_uint(f);
    return (u & 0x80000000u) ? ~u : (u | 0x80000000u);
}
__device__ __forceinline__ float key2f(unsigned k) {
    unsigned u = (k & 0x80000000u) ? (k & 0x7fffffffu) : ~k;
    return __uint_as_float(u);
}
__device__ __forceinline__ float logsig(float x) {
    return fminf(x, 0.f) - log1pf(expf(-fabsf(x)));
}

// ---------------------------------------------------------------- packing
// A-fragment layout (16-bit A 16x32, wave32): lane l holds row m=l&15,
// khalf base kb = (l<16)?0:8 ; element e -> k = (e<8) ? kb+e : kb+e+8.
// xpack fragment f = mt*8 + kt, 512 halfs, [lane][e] contiguous.
__global__ void k_pack_x(const float* __restrict__ x, _Float16* __restrict__ xp) {
    int idx = blockIdx.x * blockDim.x + threadIdx.x;
    if (idx >= NP * DD) return;
    int f = idx >> 9, r = idx & 511;
    int l = r >> 4, e = r & 15;
    int mt = f >> 3, kt = f & 7;
    int m  = mt * 16 + (l & 15);
    int kb = (l < 16) ? 0 : 8;
    int k  = kt * 32 + ((e < 8) ? (kb + e) : (kb + e + 8));
    xp[idx] = (_Float16)x[(size_t)m * DD + k];
}

// B-fragment layout (16-bit B 32x16, wave32): lane l holds col n=l&15,
// kb = (l<16)?0:8 ; element e -> k = (e<8) ? kb+e : kb+e+8.   B[k][n] = W[n][k] (h = x @ W^T)
// wpack fragment f = kt*16 + nt, 512 halfs, [lane][e] contiguous.
__global__ void k_pack_w(const float* __restrict__ w, _Float16* __restrict__ wp) {
    int idx = blockIdx.x * blockDim.x + threadIdx.x;
    if (idx >= DD * DD) return;
    int f = idx >> 9, r = idx & 511;
    int l = r >> 4, e = r & 15;
    int kt = f >> 4, nt = f & 15;
    int k  = kt * 32 + ((l < 16) ? 0 : 8) + ((e < 8) ? e : e + 8);
    int n  = nt * 16 + (l & 15);
    wp[idx] = (_Float16)w[(size_t)n * DD + k];
}

// ---------------------------------------------------------------- WMMA GEMM
// h[40000x256] = x @ W^T ; one wave computes one 16x16 tile, K=256 in 8 steps.
// grid (2500, 2), block 256 (8 waves): wave w -> nt = blockIdx.y*8 + w.
__global__ void k_gemm_wmma(const _Float16* __restrict__ xp,
                            const _Float16* __restrict__ wp,
                            float* __restrict__ h) {
    const int lane = threadIdx.x & 31;
    const int wv   = threadIdx.x >> 5;
    const int mt   = blockIdx.x;
    const int nt   = blockIdx.y * 8 + wv;

    v8f acc = {};
    const _Float16* ap = xp + ((size_t)mt * 8) * 512 + lane * 16;
    const _Float16* bp = wp + (size_t)nt * 512 + lane * 16;
#pragma unroll
    for (int kt = 0; kt < 8; ++kt) {
        v16h a = *(const v16h*)ap;
        v16h b = *(const v16h*)bp;
        acc = __builtin_amdgcn_wmma_f32_16x16x32_f16(
            false, a, false, b, (short)0, acc, false, false);
        ap += 512;
        bp += 16 * 512;
    }
    // C/D layout: VGPR r -> row M = r + 8*(lane>=16), col N = lane&15
    const int row0 = mt * 16 + ((lane >= 16) ? 8 : 0);
    const int col  = nt * 16 + (lane & 15);
#pragma unroll
    for (int r = 0; r < 8; ++r)
        h[(size_t)(row0 + r) * DD + col] = acc[r];
}

// ---------------------------------------------------------------- scores
// one wave per paper: three 256-d dots over the h_paper row
__global__ void k_scores_paper(const float* __restrict__ h,
                               const float* __restrict__ wl, const float* __restrict__ bl,
                               const float* __restrict__ wr, const float* __restrict__ br,
                               const float* __restrict__ ww, const float* __restrict__ bw,
                               float* __restrict__ sl, float* __restrict__ sr,
                               float* __restrict__ sw) {
    int p = (blockIdx.x * blockDim.x + threadIdx.x) >> 5;
    int lane = threadIdx.x & 31;
    if (p >= NP) return;
    const float4* hp = (const float4*)(h + (size_t)p * DD) + lane * 2;
    float4 h0 = hp[0], h1 = hp[1];
    const float4* l4 = (const float4*)wl + lane * 2;
    const float4* r4 = (const float4*)wr + lane * 2;
    const float4* w4 = (const float4*)ww + lane * 2;
    float p0 = dot4f(h0, l4[0]) + dot4f(h1, l4[1]);
    float p1 = dot4f(h0, r4[0]) + dot4f(h1, r4[1]);
    float p2 = dot4f(h0, w4[0]) + dot4f(h1, w4[1]);
    p0 = wredsum(p0); p1 = wredsum(p1); p2 = wredsum(p2);
    if (lane == 0) {
        sl[p] = p0 + bl[0];
        sr[p] = p1 + br[0];
        sw[p] = p2 + bw[0];
    }
}

// one wave per author: h_author = emb[idx]; sl_w dot
__global__ void k_scores_author(const float* __restrict__ emb,
                                const int* __restrict__ idx,
                                const float* __restrict__ wl, const float* __restrict__ bl,
                                float* __restrict__ sl) {
    int a = (blockIdx.x * blockDim.x + threadIdx.x) >> 5;
    int lane = threadIdx.x & 31;
    if (a >= NA) return;
    const float4* hp = (const float4*)(emb + (size_t)idx[a] * DD) + lane * 2;
    float4 h0 = hp[0], h1 = hp[1];
    const float4* l4 = (const float4*)wl + lane * 2;
    float p0 = dot4f(h0, l4[0]) + dot4f(h1, l4[1]);
    p0 = wredsum(p0);
    if (lane == 0) sl[a] = p0 + bl[0];
}

// ---------------------------------------------------------------- segment softmax
__global__ void k_edge_max(const int* __restrict__ ei, const int* __restrict__ ej,
                           const float* __restrict__ sl, const float* __restrict__ sr,
                           unsigned* __restrict__ mkey, int E) {
    int e = blockIdx.x * blockDim.x + threadIdx.x;
    if (e >= E) return;
    int i = ei[e], j = ej[e];
    atomicMax(mkey + i, f2key(sr[j] + sl[i]));
}

__global__ void k_edge_exp(const int* __restrict__ ei, const int* __restrict__ ej,
                           const float* __restrict__ sl, const float* __restrict__ sr,
                           const unsigned* __restrict__ mkey,
                           float* __restrict__ ecoef, float* __restrict__ z, int E) {
    int e = blockIdx.x * blockDim.x + threadIdx.x;
    if (e >= E) return;
    int i = ei[e], j = ej[e];
    float ex = expf(sr[j] + sl[i] - key2f(mkey[i]));
    ecoef[e] = ex;
    unsafeAtomicAdd(z + i, ex);
}

// one wave per edge: 256-float gather, scale, atomic scatter-add into out
__global__ void k_edge_agg(const int* __restrict__ ei, const int* __restrict__ ej,
                           const float* __restrict__ ecoef, const float* __restrict__ z,
                           const float* __restrict__ h, float* __restrict__ out, int E) {
    int e = (blockIdx.x * blockDim.x + threadIdx.x) >> 5;
    int lane = threadIdx.x & 31;
    if (e >= E) return;
    int i = ei[e], j = ej[e];
    float a = ecoef[e] / z[i];
    const float4* hp = (const float4*)(h + (size_t)j * DD) + lane * 2;
    float4 v0 = hp[0], v1 = hp[1];
    float* op = out + (size_t)i * DD + lane * 8;
    unsafeAtomicAdd(op + 0, v0.x * a);
    unsafeAtomicAdd(op + 1, v0.y * a);
    unsafeAtomicAdd(op + 2, v0.z * a);
    unsafeAtomicAdd(op + 3, v0.w * a);
    unsafeAtomicAdd(op + 4, v1.x * a);
    unsafeAtomicAdd(op + 5, v1.y * a);
    unsafeAtomicAdd(op + 6, v1.z * a);
    unsafeAtomicAdd(op + 7, v1.w * a);
}

// ---------------------------------------------------------------- beta combine
// out currently holds agg; rewrite out = agg*beta0 + self*beta1 in place.
// beta from softmax over 2 logits of xrow @ convW^T + convb.
__global__ void k_combine(const float* __restrict__ xbeta,   // rows feeding conv logits
                          const int* __restrict__ idx,       // row gather (or nullptr)
                          const float* __restrict__ convW, const float* __restrict__ convb,
                          const float* __restrict__ hself,   // self features (or == xbeta)
                          float* __restrict__ out, int N) {
    int n = (blockIdx.x * blockDim.x + threadIdx.x) >> 5;
    int lane = threadIdx.x & 31;
    if (n >= N) return;
    int row = idx ? idx[n] : n;
    const float4* xp = (const float4*)(xbeta + (size_t)row * DD) + lane * 2;
    float4 x0 = xp[0], x1 = xp[1];
    const float4* w0 = (const float4*)(convW) + lane * 2;
    const float4* w1 = (const float4*)(convW + DD) + lane * 2;
    float d0 = dot4f(x0, w0[0]) + dot4f(x1, w0[1]);
    float d1 = dot4f(x0, w1[0]) + dot4f(x1, w1[1]);
    d0 = wredsum(d0) + convb[0];
    d1 = wredsum(d1) + convb[1];
    float m  = fmaxf(d0, d1);
    float e0 = expf(d0 - m), e1 = expf(d1 - m);
    float inv = 1.f / (e0 + e1);
    float b0 = e0 * inv, b1 = e1 * inv;

    const float* hs = (idx ? (hself + (size_t)row * DD) : (hself + (size_t)n * DD));
    float4 s0 = ((const float4*)hs)[lane * 2 + 0];
    float4 s1 = ((const float4*)hs)[lane * 2 + 1];
    float4* op = (float4*)(out + (size_t)n * DD) + lane * 2;
    float4 a0 = op[0], a1 = op[1];
    a0.x = a0.x * b0 + s0.x * b1;  a0.y = a0.y * b0 + s0.y * b1;
    a0.z = a0.z * b0 + s0.z * b1;  a0.w = a0.w * b0 + s0.w * b1;
    a1.x = a1.x * b0 + s1.x * b1;  a1.y = a1.y * b0 + s1.y * b1;
    a1.z = a1.z * b0 + s1.z * b1;  a1.w = a1.w * b0 + s1.w * b1;
    op[0] = a0;
    op[1] = a1;
}

// ---------------------------------------------------------------- proximity loss
__global__ void k_loss(const float* __restrict__ sl, const float* __restrict__ sr,
                       const int* __restrict__ ei, const int* __restrict__ ej,
                       int E, float sign, float scale, float* __restrict__ loss) {
    int t = blockIdx.x * blockDim.x + threadIdx.x;
    float v = 0.f;
    if (t < E) {
        float x = sign * (sl[ei[t]] + sr[ej[t]]);
        v = scale * logsig(x);
    }
#pragma unroll
    for (int o = 16; o > 0; o >>= 1) v += __shfl_down(v, o, 32);
    if ((threadIdx.x & 31) == 0) unsafeAtomicAdd(loss, v);
}

// ---------------------------------------------------------------- launcher
extern "C" void kernel_launch(void* const* d_in, const int* in_sizes, int n_in,
                              void* d_out, int out_size, void* d_ws, size_t ws_size,
                              hipStream_t stream) {
    const float* x_paper  = (const float*)d_in[0];
    const float* W_lin    = (const float*)d_in[1];
    const float* emb_auth = (const float*)d_in[2];
    const float* convW_p  = (const float*)d_in[3];
    const float* convb_p  = (const float*)d_in[4];
    const float* convW_a  = (const float*)d_in[5];
    const float* convb_a  = (const float*)d_in[6];
    const float* Lc = (const float*)d_in[7];  const float* bLc = (const float*)d_in[8];
    const float* Rc = (const float*)d_in[9];  const float* bRc = (const float*)d_in[10];
    const float* Lw = (const float*)d_in[11]; const float* bLw = (const float*)d_in[12];
    const float* Rw = (const float*)d_in[13]; const float* bRw = (const float*)d_in[14];
    const int* ec  = (const int*)d_in[15];
    const int* ew  = (const int*)d_in[16];
    const int* ngc = (const int*)d_in[17];
    const int* ngw = (const int*)d_in[18];
    const int* xia = (const int*)d_in[19];

    // ---- workspace carve (256B aligned)
    char* p = (char*)d_ws;
    auto carve = [&](size_t bytes) {
        void* r = (void*)p;
        p += (bytes + 255) & ~(size_t)255;
        return r;
    };
    _Float16* xpack = (_Float16*)carve((size_t)NP * DD * 2);
    _Float16* wpack = (_Float16*)carve((size_t)DD * DD * 2);
    float* h     = (float*)carve((size_t)NP * DD * 4);
    float* sl_c  = (float*)carve(NP * 4);
    float* sr_c  = (float*)carve(NP * 4);
    float* sr_w  = (float*)carve(NP * 4);
    float* sl_w  = (float*)carve(NA * 4);
    unsigned* mk_c = (unsigned*)carve(NP * 4);
    float* z_c   = (float*)carve(NP * 4);
    unsigned* mk_w = (unsigned*)carve(NA * 4);
    float* z_w   = (float*)carve(NA * 4);
    float* ecf_c = (float*)carve(EC * 4);
    float* ecf_w = (float*)carve(EW * 4);

    float* out_paper = (float*)d_out;                       // [NP, 256]
    float* out_auth  = out_paper + (size_t)NP * DD;         // [NA, 256]
    float* loss      = out_auth + (size_t)NA * DD;          // [1]

    // ---- init (graph-capture safe)
    hipMemsetAsync(d_out, 0, (size_t)out_size * 4, stream);
    hipMemsetAsync(mk_c, 0, NP * 4, stream);
    hipMemsetAsync(z_c, 0, NP * 4, stream);
    hipMemsetAsync(mk_w, 0, NA * 4, stream);
    hipMemsetAsync(z_w, 0, NA * 4, stream);

    // ---- pack + GEMM (WMMA f16 -> f32)
    k_pack_x<<<(NP * DD) / 256, 256, 0, stream>>>(x_paper, xpack);
    k_pack_w<<<(DD * DD) / 256, 256, 0, stream>>>(W_lin, wpack);
    k_gemm_wmma<<<dim3(NP / 16, 2), 256, 0, stream>>>(xpack, wpack, h);

    // ---- attention scores
    k_scores_paper<<<NP / 8, 256, 0, stream>>>(h, Lc, bLc, Rc, bRc, Rw, bRw,
                                               sl_c, sr_c, sr_w);
    k_scores_author<<<NA / 8, 256, 0, stream>>>(emb_auth, xia, Lw, bLw, sl_w);

    // ---- segment softmax (cites: paper->paper, writes: paper->author)
    const int *ec_i = ec, *ec_j = ec + EC;
    const int *ew_i = ew, *ew_j = ew + EW;
    k_edge_max<<<(EC + 255) / 256, 256, 0, stream>>>(ec_i, ec_j, sl_c, sr_c, mk_c, EC);
    k_edge_max<<<(EW + 255) / 256, 256, 0, stream>>>(ew_i, ew_j, sl_w, sr_w, mk_w, EW);
    k_edge_exp<<<(EC + 255) / 256, 256, 0, stream>>>(ec_i, ec_j, sl_c, sr_c, mk_c,
                                                     ecf_c, z_c, EC);
    k_edge_exp<<<(EW + 255) / 256, 256, 0, stream>>>(ew_i, ew_j, sl_w, sr_w, mk_w,
                                                     ecf_w, z_w, EW);

    // ---- weighted scatter-aggregate directly into d_out
    k_edge_agg<<<(EC + 7) / 8, 256, 0, stream>>>(ec_i, ec_j, ecf_c, z_c, h, out_paper, EC);
    k_edge_agg<<<(EW + 7) / 8, 256, 0, stream>>>(ew_i, ew_j, ecf_w, z_w, h, out_auth, EW);

    // ---- beta combine (in-place on d_out)
    k_combine<<<NP / 8, 256, 0, stream>>>(x_paper, nullptr, convW_p, convb_p,
                                          h, out_paper, NP);
    k_combine<<<NA / 8, 256, 0, stream>>>(emb_auth, xia, convW_a, convb_a,
                                          emb_auth, out_auth, NA);

    // ---- proximity loss
    k_loss<<<(EC + 255) / 256, 256, 0, stream>>>(sl_c, sr_c, ec_i, ec_j, EC,
                                                 1.f, -1.f / EC, loss);
    k_loss<<<(EW + 255) / 256, 256, 0, stream>>>(sl_w, sr_w, ew_i, ew_j, EW,
                                                 1.f, -1.f / EW, loss);
    k_loss<<<(EC + 255) / 256, 256, 0, stream>>>(sl_c, sr_c, ngc, ngc + EC, EC,
                                                 -1.f, -1.f / EC, loss);
    k_loss<<<(EW + 255) / 256, 256, 0, stream>>>(sl_w, sr_w, ngw, ngw + EW, EW,
                                                 -1.f, -1.f / EW, loss);
}
